// CrossReplicaAttention_9637906612364
// MI455X (gfx1250) — matrix operand order repaired and checked by reading, not compiled, not run on hardware
//
#include <hip/hip_runtime.h>
#include <math.h>

// ---------------------------------------------------------------------------
// CDNA5 / gfx1250 types
// ---------------------------------------------------------------------------
typedef __attribute__((ext_vector_type(16))) __bf16        v16bf;
typedef __attribute__((ext_vector_type(8)))  float         v8f;
typedef __attribute__((ext_vector_type(4)))  unsigned int  v4u;
typedef __attribute__((ext_vector_type(8)))  unsigned int  v8u;

// Problem constants (from the reference)
#define PP   4
#define BB   2
#define SS   2048
#define DD   2048
#define HH   16
#define HKK  4
#define DHH  128
#define MROWS (PP*BB*SS)          // 16384
#define NQ   (HH*DHH)             // 2048
#define NKV  (HKK*DHH)            // 512

// ---------------------------------------------------------------------------
// fp32 -> bf16 split helpers (round-to-nearest-even)
// ---------------------------------------------------------------------------
__device__ __forceinline__ unsigned short bf16_rne(float x) {
    union { float f; unsigned u; } v; v.f = x;
    unsigned u = v.u;
    unsigned r = u + 0x7FFFu + ((u >> 16) & 1u);
    return (unsigned short)(r >> 16);
}
__device__ __forceinline__ float bf16_to_f(unsigned short h) {
    union { unsigned u; float f; } v; v.u = ((unsigned)h) << 16;
    return v.f;
}

// Generic LDS pointers carry the wave-relative LDS byte offset in addr[31:0]
// (ISA 10.2 aperture mapping), which is what the TDM descriptor wants.
__device__ __forceinline__ unsigned lds_off(const void* p) {
    return (unsigned)(unsigned long long)p;
}

// ---------------------------------------------------------------------------
// Tensor Data Mover: async 2-D tile load global -> LDS.
// Builds D# group0 (128b) + group1 (256b) per ISA 8.3/8.4 and issues the
// 2-group form of tensor_load_to_lds (tensors up to 2D, VADDR2/3 = NULL).
//   tile_d0 = contiguous elements per row, tile_d1 = rows,
//   stride  = row stride in elements, data_size = 4 bytes (fp32).
// Must be executed by a single wave with uniform operands.
// ---------------------------------------------------------------------------
__device__ __forceinline__ void tdm_load_2d(unsigned lds_byte_off,
                                            const float* gptr,
                                            unsigned tile_d0, unsigned tile_d1,
                                            unsigned stride_elems)
{
    unsigned long long ga = (unsigned long long)gptr;
    v4u g0;
    g0[0] = 1u;                                           // count=1, user mode
    g0[1] = lds_byte_off;                                 // lds_addr
    g0[2] = (unsigned)(ga & 0xFFFFFFFFu);                 // global_addr[31:0]
    g0[3] = (unsigned)((ga >> 32) & 0x01FFFFFFu)          // global_addr[56:32]
          | (2u << 30);                                   // type = 2 ("image")
    v8u g1;
    g1[0] = 2u << 16;                                     // data_size = 4B
    g1[1] = (tile_d0 & 0xFFFFu) << 16;                    // tensor_dim0[15:0]
    g1[2] = (tile_d1 & 0xFFFFu) << 16;                    // tensor_dim1[15:0]
    g1[3] = (tile_d0 & 0xFFFFu) << 16;                    // tile_dim0
    g1[4] = (tile_d1 & 0xFFFFu);                          // tile_dim1 (tile_dim2=0)
    g1[5] = stride_elems;                                 // tensor_dim0_stride[31:0]
    g1[6] = 0u;                                           // stride hi / dim1_stride
    g1[7] = 0u;
    asm volatile("tensor_load_to_lds %0, %1" :: "s"(g0), "s"(g1) : "memory");
}

// ---------------------------------------------------------------------------
// 128x128 block-tile GEMM, K in steps of 32, bf16 split-precision WMMA.
//   out[m0..m0+128, wcol0..wcol0+128) = A[m0.., :K] * W[:K, wcol0..) + bias
// Pipeline per K-step:
//   TDM (wave 0): global -> LDS raw fp32 tile for step kt+1  (TENSORcnt)
//   all waves   : LDS raw (kt) -> bf16 hi/lo limbs, swizzled into WMMA
//                 fragment order in LDS
//   all waves   : v_wmma_f32_16x16x32_bf16 x3 per fragment (split precision)
// ---------------------------------------------------------------------------
__device__ __forceinline__ void gemm_tile_128x128(
    const float* __restrict__ A, int lda,
    const float* __restrict__ W, int ldw, int wcol0,
    const float* __restrict__ bias,
    float* __restrict__ out, int ldo,
    int m0, int K)
{
    __shared__ __align__(16) float Araw[2][128 * 32];     // 2 x 16 KB
    __shared__ __align__(16) float Braw[2][32 * 128];     // 2 x 16 KB
    __shared__ __align__(32) unsigned short Ahi[8][32][16];
    __shared__ __align__(32) unsigned short Alo[8][32][16];
    __shared__ __align__(32) unsigned short Bhi[8][32][16];
    __shared__ __align__(32) unsigned short Blo[8][32][16];

    const int tid  = threadIdx.x;
    const int lane = tid & 31;
    const int wave = tid >> 5;
    const int wm   = wave >> 1;   // 0..3 (M direction, 2 tiles each)
    const int wn   = wave & 1;    // 0..1 (N direction, 4 tiles each)

    const v8f vzero = {0.f,0.f,0.f,0.f,0.f,0.f,0.f,0.f};
    v8f acc[2][4];
#pragma unroll
    for (int mi = 0; mi < 2; ++mi)
#pragma unroll
        for (int ni = 0; ni < 4; ++ni) acc[mi][ni] = vzero;

    const int KT = K >> 5;   // K / 32

    // Prologue: wave 0 kicks off the TDM for K-step 0.
    if (tid < 32) {
        tdm_load_2d(lds_off(&Araw[0][0]), A + (size_t)m0 * lda,      32, 128, (unsigned)lda);
        tdm_load_2d(lds_off(&Braw[0][0]), W + wcol0,                128,  32, (unsigned)ldw);
    }

    for (int kt = 0; kt < KT; ++kt) {
        const int buf = kt & 1;

        if (tid < 32) {
            if (kt + 1 < KT) {
                const int k1 = (kt + 1) << 5;
                // issue next tile's DMA, then wait for the current tile's two
                // (older) transfers: TENSORcnt completes in order.
                tdm_load_2d(lds_off(&Araw[buf ^ 1][0]),
                            A + (size_t)m0 * lda + k1, 32, 128, (unsigned)lda);
                tdm_load_2d(lds_off(&Braw[buf ^ 1][0]),
                            W + (size_t)k1 * ldw + wcol0, 128, 32, (unsigned)ldw);
                __builtin_amdgcn_s_wait_tensorcnt(2);
            } else {
                __builtin_amdgcn_s_wait_tensorcnt(0);
            }
        }
        __syncthreads();   // raw tile kt visible; previous frag reads drained

        // ---- convert raw fp32 tile -> bf16 hi/lo limbs in WMMA fragment order
#pragma unroll
        for (int i = 0; i < 4; ++i) {
            int quad = tid + i * 256;
            {   // A: 128 rows x 32 k; 16-bit A-matrix 16x32 layout (ISA 7.12.2)
                int row = quad >> 3;             // 0..127
                int kq  = (quad & 7) << 2;       // 0..28
                const float4 a4 = *(const float4*)&Araw[buf][row * 32 + kq];
                float av[4] = {a4.x, a4.y, a4.z, a4.w};
                int mt = row >> 4, r = row & 15;
#pragma unroll
                for (int j = 0; j < 4; ++j) {
                    int k = kq + j;
                    float x = av[j];
                    unsigned short h = bf16_rne(x);
                    unsigned short l = bf16_rne(x - bf16_to_f(h));
                    int ln = r + ((k & 8) ? 16 : 0);
                    int pr = (k < 16) ? ((k & 7) >> 1) : (4 + ((k & 7) >> 1));
                    int el = (pr << 1) + (k & 1);
                    Ahi[mt][ln][el] = h;
                    Alo[mt][ln][el] = l;
                }
            }
            {   // B: 32 k-rows x 128 cols; lanes 0-15 K=0..7,16..23, lanes 16-31 K=8..15,24..31
                int krow = quad >> 5;            // 0..31
                int nq   = (quad & 31) << 2;     // 0..124
                const float4 b4 = *(const float4*)&Braw[buf][krow * 128 + nq];
                float bvv[4] = {b4.x, b4.y, b4.z, b4.w};
#pragma unroll
                for (int j = 0; j < 4; ++j) {
                    int n = nq + j;
                    float x = bvv[j];
                    unsigned short h = bf16_rne(x);
                    unsigned short l = bf16_rne(x - bf16_to_f(h));
                    int nt = n >> 4;
                    int ln = (n & 15) + ((krow & 8) ? 16 : 0);
                    int el = (krow & 7) + ((krow >= 16) ? 8 : 0);
                    Bhi[nt][ln][el] = h;
                    Blo[nt][ln][el] = l;
                }
            }
        }

        __syncthreads();   // fragments visible

        // ---- fragment loads (one 32B LDS read each) + WMMA
        v16bf ah[2], al[2], bh[4], bl[4];
#pragma unroll
        for (int mi = 0; mi < 2; ++mi) {
            int mt = wm * 2 + mi;
            ah[mi] = *reinterpret_cast<const v16bf*>(&Ahi[mt][lane][0]);
            al[mi] = *reinterpret_cast<const v16bf*>(&Alo[mt][lane][0]);
        }
#pragma unroll
        for (int ni = 0; ni < 4; ++ni) {
            int nt = wn * 4 + ni;
            bh[ni] = *reinterpret_cast<const v16bf*>(&Bhi[nt][lane][0]);
            bl[ni] = *reinterpret_cast<const v16bf*>(&Blo[nt][lane][0]);
        }
#pragma unroll
        for (int mi = 0; mi < 2; ++mi) {
#pragma unroll
            for (int ni = 0; ni < 4; ++ni) {
                // split precision: Ahi*Bhi + Ahi*Blo + Alo*Bhi, fp32 accum
                acc[mi][ni] = __builtin_amdgcn_wmma_f32_16x16x32_bf16(
                    false, ah[mi], false, bh[ni], (short)0, acc[mi][ni], false, false);
                acc[mi][ni] = __builtin_amdgcn_wmma_f32_16x16x32_bf16(
                    false, ah[mi], false, bl[ni], (short)0, acc[mi][ni], false, false);
                acc[mi][ni] = __builtin_amdgcn_wmma_f32_16x16x32_bf16(
                    false, al[mi], false, bh[ni], (short)0, acc[mi][ni], false, false);
            }
        }
    }

    // ---- epilogue: C layout (ISA 7.12.2): VGPR r -> M = r + (lane>=16?8:0), N = lane%16
    const int rbase = (lane >= 16) ? 8 : 0;
    const int ncol  = lane & 15;
#pragma unroll
    for (int mi = 0; mi < 2; ++mi) {
#pragma unroll
        for (int ni = 0; ni < 4; ++ni) {
            int ocol = wcol0 + (wn * 4 + ni) * 16 + ncol;
            float bv = bias ? bias[ocol] : 0.0f;
#pragma unroll
            for (int r = 0; r < 8; ++r) {
                int m = m0 + (wm * 2 + mi) * 16 + rbase + r;
                out[(size_t)m * ldo + ocol] = acc[mi][ni][r] + bv;
            }
        }
    }
}

// ---------------------------------------------------------------------------
// Kernel 1: fused QKV projection.  N-space = [q:0..2048) [k:2048..2560) [v:..3072)
// ---------------------------------------------------------------------------
__global__ void __launch_bounds__(256)
qkv_gemm_kernel(const float* __restrict__ hs,
                const float* __restrict__ Wq, const float* __restrict__ bq,
                const float* __restrict__ Wk, const float* __restrict__ bk,
                const float* __restrict__ Wv, const float* __restrict__ bv,
                float* __restrict__ q_ws, float* __restrict__ k_ws,
                float* __restrict__ v_ws)
{
    const int n0 = blockIdx.x * 128;        // 0..3071
    const int m0 = blockIdx.y * 128;        // 0..16383

    const float* W; const float* bias; float* out;
    int ldw, ldo, wcol0;
    if (n0 < NQ) {
        W = Wq; bias = bq; out = q_ws; ldw = NQ;  ldo = NQ;  wcol0 = n0;
    } else if (n0 < NQ + NKV) {
        W = Wk; bias = bk; out = k_ws; ldw = NKV; ldo = NKV; wcol0 = n0 - NQ;
    } else {
        W = Wv; bias = bv; out = v_ws; ldw = NKV; ldo = NKV; wcol0 = n0 - NQ - NKV;
    }
    gemm_tile_128x128(hs, DD, W, ldw, wcol0, bias, out, ldo, m0, DD);
}

// ---------------------------------------------------------------------------
// Kernel 2: RoPE + replica attention (P=4 keys per query), fp32 VALU.
// One thread per (b,s,h,p).  b is constant within a 256-thread block, so
// cos/sin for all 4 replicas are staged in LDS once.
// ---------------------------------------------------------------------------
__global__ void __launch_bounds__(256)
replica_attn_kernel(const float* __restrict__ q_ws,
                    const float* __restrict__ k_ws,
                    const float* __restrict__ v_ws,
                    const float* __restrict__ cosT,
                    const float* __restrict__ sinT,
                    float* __restrict__ o_ws)
{
    __shared__ float cs[PP][DHH];
    __shared__ float sn[PP][DHH];

    const int  tid = threadIdx.x;
    const long idx = (long)blockIdx.x * 256 + tid;   // (b,s,h,p) flattened
    const int  b   = (int)(idx >> 17);               // constant per block

    for (int i = tid; i < PP * DHH; i += 256) {
        int j = i >> 7, d = i & 127;
        cs[j][d] = cosT[(size_t)(b * PP + j) * DHH + d];
        sn[j][d] = sinT[(size_t)(b * PP + j) * DHH + d];
    }
    __syncthreads();

    const int p  = (int)(idx & 3);
    const int h  = (int)((idx >> 2) & 15);
    const int s  = (int)((idx >> 6) & 2047);
    const int hk = h & (HKK - 1);                    // tile: k[h % HK]

    const size_t qrow = (size_t)(p * BB + b) * SS + s;
    const float* qp = q_ws + qrow * NQ + h * DHH;

    const float* kb[PP];
    const float* vb[PP];
#pragma unroll
    for (int j = 0; j < PP; ++j) {
        size_t r = (size_t)(j * BB + b) * SS + s;
        kb[j] = k_ws + r * NKV + hk * DHH;
        vb[j] = v_ws + r * NKV + hk * DHH;
    }

    // Pass 1: RoPE(q) . RoPE(k_j) for the 4 replicas
    float sc[PP] = {0.f, 0.f, 0.f, 0.f};
    for (int d = 0; d < DHH / 2; ++d) {
        float q0 = qp[d], q1 = qp[d + 64];
        float qr0 = q0 * cs[p][d]      - q1 * sn[p][d];
        float qr1 = q1 * cs[p][d + 64] + q0 * sn[p][d + 64];
#pragma unroll
        for (int j = 0; j < PP; ++j) {
            float k0 = kb[j][d], k1 = kb[j][d + 64];
            float kr0 = k0 * cs[j][d]      - k1 * sn[j][d];
            float kr1 = k1 * cs[j][d + 64] + k0 * sn[j][d + 64];
            sc[j] += qr0 * kr0 + qr1 * kr1;
        }
    }

    // softmax over the 4 replicas (scaled by DH^-0.5)
    const float scale = 0.08838834764831845f;  // 1/sqrt(128)
    float m = fmaxf(fmaxf(sc[0], sc[1]), fmaxf(sc[2], sc[3])) * scale;
    float w[PP], sum = 0.f;
#pragma unroll
    for (int j = 0; j < PP; ++j) { w[j] = __expf(sc[j] * scale - m); sum += w[j]; }
    float inv = 1.0f / sum;
#pragma unroll
    for (int j = 0; j < PP; ++j) w[j] *= inv;

    // Pass 2: o = sum_j w_j * v_j, stored in (P*B, S, H*DH) layout for Wo GEMM
    float* op = o_ws + qrow * NQ + h * DHH;
    for (int d = 0; d < DHH; d += 4) {
        float4 v0 = *(const float4*)(vb[0] + d);
        float4 v1 = *(const float4*)(vb[1] + d);
        float4 v2 = *(const float4*)(vb[2] + d);
        float4 v3 = *(const float4*)(vb[3] + d);
        float4 o;
        o.x = w[0] * v0.x + w[1] * v1.x + w[2] * v2.x + w[3] * v3.x;
        o.y = w[0] * v0.y + w[1] * v1.y + w[2] * v2.y + w[3] * v3.y;
        o.z = w[0] * v0.z + w[1] * v1.z + w[2] * v2.z + w[3] * v3.z;
        o.w = w[0] * v0.w + w[1] * v1.w + w[2] * v2.w + w[3] * v3.w;
        *(float4*)(op + d) = o;
    }
}

// ---------------------------------------------------------------------------
// Kernel 3: output projection o @ Wo -> d_out
// ---------------------------------------------------------------------------
__global__ void __launch_bounds__(256)
out_gemm_kernel(const float* __restrict__ o_ws,
                const float* __restrict__ Wo,
                float* __restrict__ out)
{
    const int n0 = blockIdx.x * 128;
    const int m0 = blockIdx.y * 128;
    gemm_tile_128x128(o_ws, NQ, Wo, DD, n0, nullptr, out, DD, m0, NQ);
}

// ---------------------------------------------------------------------------
// Host-side launcher
// ---------------------------------------------------------------------------
extern "C" void kernel_launch(void* const* d_in, const int* in_sizes, int n_in,
                              void* d_out, int out_size, void* d_ws, size_t ws_size,
                              hipStream_t stream)
{
    const float* hs   = (const float*)d_in[0];
    const float* cosT = (const float*)d_in[1];
    const float* sinT = (const float*)d_in[2];
    const float* Wq   = (const float*)d_in[3];
    const float* bq   = (const float*)d_in[4];
    const float* Wk   = (const float*)d_in[5];
    const float* bk   = (const float*)d_in[6];
    const float* Wv   = (const float*)d_in[7];
    const float* bv   = (const float*)d_in[8];
    const float* Wo   = (const float*)d_in[9];
    float* out = (float*)d_out;

    // Workspace carve-out: q | k | v | o   (total 335 MB)
    float* q_ws = (float*)d_ws;
    float* k_ws = q_ws + (size_t)MROWS * NQ;
    float* v_ws = k_ws + (size_t)MROWS * NKV;
    float* o_ws = v_ws + (size_t)MROWS * NKV;

    // 1) QKV projection: N = 3072 (q|k|v), M = 16384
    dim3 gridQKV((NQ + 2 * NKV) / 128, MROWS / 128);
    qkv_gemm_kernel<<<gridQKV, 256, 0, stream>>>(hs, Wq, bq, Wk, bk, Wv, bv,
                                                 q_ws, k_ws, v_ws);

    // 2) RoPE + 4-way replica attention: B*S*H*P = 262144 threads
    dim3 gridAtt((BB * SS * HH * PP) / 256);
    replica_attn_kernel<<<gridAtt, 256, 0, stream>>>(q_ws, k_ws, v_ws,
                                                     cosT, sinT, o_ws);

    // 3) Output projection: 16384 x 2048 x 2048
    dim3 gridO(DD / 128, MROWS / 128);
    out_gemm_kernel<<<gridO, 256, 0, stream>>>(o_ws, Wo, out);
}
